// ConvexGenerator_49555332661956
// MI455X (gfx1250) — compile-verified, matrix operand order checked by reading, not run on hardware
//
#include <hip/hip_runtime.h>
#include <hip/hip_bf16.h>

#define BB      2048
#define LATENT  128
#define NCLS    8
#define HID     1024
#define DDIM    512
#define NMAXN   4096
#define PADB    (BB + NCLS * 32)   /* 2304: class segments aligned to 32 */
#define KPAD1   160

typedef __attribute__((ext_vector_type(8)))  float  v8f;
typedef __attribute__((ext_vector_type(16))) __bf16 v16bf;
typedef __attribute__((ext_vector_type(8)))  __bf16 v8bf;

__device__ __constant__ int KCOUNTS[8] = {1024, 1536, 2048, 2560, 3072, 3584, 3840, 4096};

__device__ __forceinline__ v16bf cat16(v8bf lo, v8bf hi) {
    return __builtin_shufflevector(lo, hi, 0,1,2,3,4,5,6,7,8,9,10,11,12,13,14,15);
}
__device__ __forceinline__ v8f wmma_bf16f32(v16bf a, v16bf b, v8f c) {
    return __builtin_amdgcn_wmma_f32_16x16x32_bf16(false, a, false, b, (short)0, c, false, false);
}
__device__ __forceinline__ float gelu_exact(float v) {
    return 0.5f * v * (1.0f + erff(v * 0.70710678118654752f));
}
// pack two fp32 -> one dword of two bf16. Inline asm pins a single
// v_cvt_pk_bf16_f32 (a -> low half, b -> high half); prevents the SLP
// vectorizer from re-pairing the converts and emitting shift/or glue.
__device__ __forceinline__ unsigned pkbf(float a, float b) {
    unsigned r;
    asm("v_cvt_pk_bf16_f32 %0, %1, %2" : "=v"(r) : "v"(a), "v"(b));
    return r;
}
// A fragment (16x32 bf16, ISA layout): lane m = L&15, K-half hb = L>>4
__device__ __forceinline__ v16bf loadA(const __bf16* row, int k0, int hb) {
    v8bf lo = *(const v8bf*)(row + k0 + hb * 8);
    v8bf hi = *(const v8bf*)(row + k0 + 16 + hb * 8);
    return cat16(lo, hi);
}
// B fragment from LDS tile laid out [n][k] (k contiguous, 32 per row)
__device__ __forceinline__ v16bf loadB(const __bf16* bt, int mn, int hb) {
    v8bf lo = *(const v8bf*)(bt + mn * 32 + hb * 8);
    v8bf hi = *(const v8bf*)(bt + mn * 32 + 16 + hb * 8);
    return cat16(lo, hi);
}

// Stage a 32(k)x16(n) fp32 tile -> LDS bf16 [n][k]; NO bounds check (hot path).
// Lane owns k rows {2*kp, 2*kp+1}, n half nh*8..+8.
__device__ __forceinline__ void stage_w_full(unsigned* bt32, const float* W, size_t ldw,
                                             int k0, int n0, int kp, int nh) {
    const float* p = W + (size_t)(k0 + 2 * kp) * ldw + n0 + nh * 8;
    const float4* a4 = (const float4*)p;
    const float4* b4 = (const float4*)(p + ldw);
    float4 a0 = a4[0], a1 = a4[1];
    float4 b0 = b4[0], b1 = b4[1];
    const int base = nh * 8;
    bt32[(base + 0) * 16 + kp] = pkbf(a0.x, b0.x);
    bt32[(base + 1) * 16 + kp] = pkbf(a0.y, b0.y);
    bt32[(base + 2) * 16 + kp] = pkbf(a0.z, b0.z);
    bt32[(base + 3) * 16 + kp] = pkbf(a0.w, b0.w);
    bt32[(base + 4) * 16 + kp] = pkbf(a1.x, b1.x);
    bt32[(base + 5) * 16 + kp] = pkbf(a1.y, b1.y);
    bt32[(base + 6) * 16 + kp] = pkbf(a1.z, b1.z);
    bt32[(base + 7) * 16 + kp] = pkbf(a1.w, b1.w);
}
// Guarded variant for the single ragged tail tile (layer 1 only).
__device__ __forceinline__ void stage_w_tail(unsigned* bt32, const float* W, size_t ldw,
                                             int Kvalid, int k0, int n0, int kp, int nh) {
    const int kg0 = k0 + 2 * kp, kg1 = kg0 + 1;
    float w0[8], w1[8];
    if (kg0 < Kvalid) {
        const float4* p = (const float4*)(W + (size_t)kg0 * ldw + n0 + nh * 8);
        float4 a = p[0], b = p[1];
        w0[0]=a.x; w0[1]=a.y; w0[2]=a.z; w0[3]=a.w; w0[4]=b.x; w0[5]=b.y; w0[6]=b.z; w0[7]=b.w;
    } else {
#pragma unroll
        for (int j = 0; j < 8; ++j) w0[j] = 0.0f;
    }
    if (kg1 < Kvalid) {
        const float4* p = (const float4*)(W + (size_t)kg1 * ldw + n0 + nh * 8);
        float4 a = p[0], b = p[1];
        w1[0]=a.x; w1[1]=a.y; w1[2]=a.z; w1[3]=a.w; w1[4]=b.x; w1[5]=b.y; w1[6]=b.z; w1[7]=b.w;
    } else {
#pragma unroll
        for (int j = 0; j < 8; ++j) w1[j] = 0.0f;
    }
#pragma unroll
    for (int j = 0; j < 8; ++j)
        bt32[(nh * 8 + j) * 16 + kp] = pkbf(w0[j], w1[j]);
}

// ---------------------------------------------------------------------------
// Kernel 1: deterministic counting sort; class segments aligned to 32, -1 pad.
// ---------------------------------------------------------------------------
__global__ __launch_bounds__(256) void k_sort(const int* __restrict__ cls,
                                              int* __restrict__ perm,
                                              int* __restrict__ abase) {
    __shared__ int cnt[256][NCLS];
    __shared__ int ctot[NCLS];
    __shared__ int cbase[NCLS + 1];
    const int tid = threadIdx.x;
    int loc[8];
#pragma unroll
    for (int c = 0; c < NCLS; ++c) cnt[tid][c] = 0;
#pragma unroll
    for (int j = 0; j < 8; ++j) {
        int c = cls[tid * 8 + j];
        loc[j] = c;
        cnt[tid][c] += 1;
    }
    __syncthreads();
    if (tid < NCLS) {
        int t = 0;
        for (int i = 0; i < 256; ++i) t += cnt[i][tid];
        ctot[tid] = t;
    }
    __syncthreads();
    if (tid == 0) {
        int run = 0;
        for (int c = 0; c < NCLS; ++c) { cbase[c] = run; run += (ctot[c] + 31) & ~31; }
        cbase[NCLS] = run;
        for (int c = 0; c <= NCLS; ++c) abase[c] = cbase[c];
    }
    __syncthreads();
    for (int i = tid; i < PADB; i += 256) perm[i] = -1;
    __syncthreads();
    int off[NCLS];
#pragma unroll
    for (int c = 0; c < NCLS; ++c) {
        int o = cbase[c];
        for (int t = 0; t < tid; ++t) o += cnt[t][c];
        off[c] = o;
    }
#pragma unroll
    for (int j = 0; j < 8; ++j) {
        int c = loc[j];
        perm[off[c]] = tid * 8 + j;
        off[c] += 1;
    }
}

// ---------------------------------------------------------------------------
// Kernel 2: xin = bf16(concat(z, onehot)) zero-padded to K=160.
// ---------------------------------------------------------------------------
__global__ __launch_bounds__(256) void k_xin(const float* __restrict__ z,
                                             const int* __restrict__ cls,
                                             unsigned short* __restrict__ xin_) {
    __bf16* xin = (__bf16*)xin_;
    int i = blockIdx.x * 256 + threadIdx.x;
    if (i >= BB * KPAD1) return;
    int b = i / KPAD1, k = i - b * KPAD1;
    float v = 0.0f;
    if (k < LATENT)             v = z[b * LATENT + k];
    else if (k < LATENT + NCLS) v = (cls[b] == (k - LATENT)) ? 1.0f : 0.0f;
    xin[i] = (__bf16)v;
}

// ---------------------------------------------------------------------------
// Kernel 3: bf16 GEMM + bias + exact GELU -> bf16. 16(M)x128(N) per block.
// Unguarded main K loop (double-buffered, 2 WMMA chains) + guarded tail.
// ---------------------------------------------------------------------------
__global__ __launch_bounds__(256) void k_mlp(const unsigned short* __restrict__ A_, int lda,
                                             const float* __restrict__ W, int N, int Kvalid,
                                             const float* __restrict__ bias,
                                             unsigned short* __restrict__ Out_, int ldo) {
    const __bf16* A = (const __bf16*)A_;
    __bf16* Out = (__bf16*)Out_;
    __shared__ __align__(16) __bf16 Bt[8][1024];   // 2 x [16][32] per wave
    const int tid = threadIdx.x;
    const int w = tid >> 5, L = tid & 31;
    const int hb = L >> 4, mn = L & 15;
    const int kp = L & 15, nh = L >> 4;
    const int m0 = blockIdx.x * 16;
    const int n0 = blockIdx.y * 128 + w * 16;
    const int Kpad  = (Kvalid + 31) & ~31;
    const int Kmain = Kvalid & ~63;
    __bf16* bt0 = &Bt[w][0];
    __bf16* bt1 = &Bt[w][512];
    const __bf16* Arow = A + (size_t)(m0 + mn) * lda;
    v8f acc0 = {}, acc1 = {};
    for (int k0 = 0; k0 < Kmain; k0 += 64) {
        stage_w_full((unsigned*)bt0, W, N, k0, n0, kp, nh);
        acc0 = wmma_bf16f32(loadA(Arow, k0, hb), loadB(bt0, mn, hb), acc0);
        stage_w_full((unsigned*)bt1, W, N, k0 + 32, n0, kp, nh);
        acc1 = wmma_bf16f32(loadA(Arow, k0 + 32, hb), loadB(bt1, mn, hb), acc1);
    }
    for (int k0 = Kmain; k0 < Kpad; k0 += 32) {     // at most one ragged tile
        stage_w_tail((unsigned*)bt0, W, N, Kvalid, k0, n0, kp, nh);
        acc0 = wmma_bf16f32(loadA(Arow, k0, hb), loadB(bt0, mn, hb), acc0);
    }
    const float bsv = bias[n0 + mn];
#pragma unroll
    for (int r = 0; r < 8; ++r) {
        int m = m0 + r + 8 * hb;
        float v = gelu_exact(acc0[r] + acc1[r] + bsv);
        Out[(size_t)m * ldo + n0 + mn] = (__bf16)v;
    }
}

// ---------------------------------------------------------------------------
// Kernel 4: fused logits + online softmax + (alpha @ Xbuf[c]).
// One block = 32 same-class samples (halves L2 traffic on Wa/Xbuf), 8 wave32.
// Each staged B tile feeds two WMMAs (row halves) -> natural 2-chain ILP.
// ---------------------------------------------------------------------------
__global__ __launch_bounds__(256) void k_attend(const unsigned short* __restrict__ T_,
                                                const float* __restrict__ Wa,
                                                const float* __restrict__ ba,
                                                const float* __restrict__ Xbuf,
                                                const int* __restrict__ perm,
                                                const int* __restrict__ abase,
                                                float* __restrict__ Out) {
    const __bf16* T = (const __bf16*)T_;
    __shared__ float Lt[32][132];                   // logits tile
    __shared__ __align__(16) __bf16 E[32][128];     // exp tile (bf16)
    __shared__ __align__(16) __bf16 uWX[8][2048];   // per-wave: Wa-stage / X-stage
    __shared__ __align__(16) float rowmax[32];
    __shared__ __align__(16) float rowsum[32];
    __shared__ __align__(16) float rowscale[32];
    __shared__ int sperm[32];
    __shared__ int scls;

    const int tid = threadIdx.x;
    const int w = tid >> 5, L = tid & 31;
    const int hb = L >> 4, mn = L & 15;
    const int kp = L & 15, nh = L >> 4;
    const int blk = blockIdx.x;

    if (tid < 32) {
        rowmax[tid] = -3.0e38f;
        rowsum[tid] = 0.0f;
        sperm[tid] = perm[blk * 32 + tid];
    }
    if (tid == 0) {
        int total = abase[NCLS];
        int c = -1;
        if (blk * 32 < total) {
            for (int cc = 0; cc < NCLS; ++cc)
                if (blk * 32 >= abase[cc] && blk * 32 < abase[cc + 1]) c = cc;
        }
        scls = c;
    }
    __syncthreads();
    const int c = scls;
    if (c < 0) return;                              // uniform across block
    const int cnt = KCOUNTS[c];

    const float* WaC = Wa + (size_t)c * HID * NMAXN;
    const float* baC = ba + (size_t)c * NMAXN;
    const float* XC  = Xbuf + (size_t)c * NMAXN * DDIM;

    const int rf0 = sperm[mn], rf1 = sperm[16 + mn];
    const __bf16* ArowLo = T + (size_t)(rf0 >= 0 ? rf0 : 0) * HID;   // dummy row if invalid
    const __bf16* ArowHi = T + (size_t)(rf1 >= 0 ? rf1 : 0) * HID;

    __bf16* Wt = &uWX[w][0];                        // phase-1 view [16 n][32 k]
    __bf16* Xt = &uWX[w][0];                        // phase-3 view [64 d][32 n]
    unsigned* xt32 = (unsigned*)Xt;

    v8f accD[8];
#pragma unroll
    for (int s = 0; s < 8; ++s) accD[s] = (v8f){};

    const int r2 = tid >> 3, cg = tid & 7;          // softmax: 8 lanes per row

    for (int n0 = 0; n0 < cnt; n0 += 128) {
        const int n0w = n0 + w * 16;
        // ---------------- phase 1: logits, 2 WMMA per staged tile ----------------
        v8f cl0 = {}, cl1 = {};
        for (int k0 = 0; k0 < HID; k0 += 32) {
            if (k0 + 32 < HID)
                __builtin_prefetch((const void*)(WaC + (size_t)(k0 + 32 + 2 * kp) * NMAXN + n0w), 0, 0);
            stage_w_full((unsigned*)Wt, WaC, NMAXN, k0, n0w, kp, nh);
            v16bf b = loadB(Wt, mn, hb);
            cl0 = wmma_bf16f32(loadA(ArowLo, k0, hb), b, cl0);
            cl1 = wmma_bf16f32(loadA(ArowHi, k0, hb), b, cl1);
        }
        const float bsv = baC[n0w + mn];
#pragma unroll
        for (int r = 0; r < 8; ++r) {
            Lt[r + 8 * hb][w * 16 + mn]      = cl0[r] + bsv;
            Lt[16 + r + 8 * hb][w * 16 + mn] = cl1[r] + bsv;
        }
        __syncthreads();
        // ---------------- phase 2: online softmax (intra-wave butterflies) -------
        float vloc[16];
        float mx = -3.0e38f;
#pragma unroll
        for (int j = 0; j < 16; ++j) {
            float v = Lt[r2][cg * 16 + j];
            vloc[j] = v;
            mx = fmaxf(mx, v);
        }
#pragma unroll
        for (int o = 1; o < 8; o <<= 1) mx = fmaxf(mx, __shfl_xor(mx, o, 32));
        const float oldm = rowmax[r2];
        const float nm = fmaxf(oldm, mx);
        float e[16];
        float ssum = 0.0f;
#pragma unroll
        for (int j = 0; j < 16; ++j) {
            e[j] = __expf(vloc[j] - nm);
            ssum += e[j];
        }
        unsigned* ep = (unsigned*)(&E[r2][cg * 16]);
#pragma unroll
        for (int jj = 0; jj < 8; ++jj) ep[jj] = pkbf(e[2 * jj], e[2 * jj + 1]);
#pragma unroll
        for (int o = 1; o < 8; o <<= 1) ssum += __shfl_xor(ssum, o, 32);
        if (cg == 0) {
            float sc = __expf(oldm - nm);
            rowscale[r2] = sc;
            rowmax[r2] = nm;
            rowsum[r2] = rowsum[r2] * sc + ssum;
        }
        __syncthreads();
        // ---------------- phase 3: alpha @ Xbuf ----------------
        const float4* rlo = (const float4*)&rowscale[hb * 8];
        const float4* rhi = (const float4*)&rowscale[16 + hb * 8];
        float4 f0 = rlo[0], f1 = rlo[1], g0 = rhi[0], g1 = rhi[1];
        float fscLo[8] = {f0.x, f0.y, f0.z, f0.w, f1.x, f1.y, f1.z, f1.w};
        float fscHi[8] = {g0.x, g0.y, g0.z, g0.w, g1.x, g1.y, g1.z, g1.w};
#pragma unroll
        for (int s = 0; s < 4; ++s)
#pragma unroll
            for (int r = 0; r < 8; ++r) {
                accD[s][r]     *= fscLo[r];
                accD[s + 4][r] *= fscHi[r];
            }
        for (int q = 0; q < 4; ++q) {
            // stage Xbuf [32 n][64 d] -> Xt [d][n] bf16 (packed dword stores)
            const float4* xp0 = (const float4*)(XC + (size_t)(n0 + q * 32 + 2 * kp) * DDIM + w * 64 + nh * 32);
            const float4* xp1 = (const float4*)(XC + (size_t)(n0 + q * 32 + 2 * kp + 1) * DDIM + w * 64 + nh * 32);
#pragma unroll
            for (int t4 = 0; t4 < 8; ++t4) {
                float4 a = xp0[t4], b = xp1[t4];
                int dd = nh * 32 + t4 * 4;
                xt32[(dd + 0) * 16 + kp] = pkbf(a.x, b.x);
                xt32[(dd + 1) * 16 + kp] = pkbf(a.y, b.y);
                xt32[(dd + 2) * 16 + kp] = pkbf(a.z, b.z);
                xt32[(dd + 3) * 16 + kp] = pkbf(a.w, b.w);
            }
            v16bf a0 = cat16(*(const v8bf*)(&E[mn][q * 32 + hb * 8]),
                             *(const v8bf*)(&E[mn][q * 32 + 16 + hb * 8]));
            v16bf a1 = cat16(*(const v8bf*)(&E[16 + mn][q * 32 + hb * 8]),
                             *(const v8bf*)(&E[16 + mn][q * 32 + 16 + hb * 8]));
#pragma unroll
            for (int s = 0; s < 4; ++s) {
                v16bf b = loadB(Xt, s * 16 + mn, hb);
                accD[s]     = wmma_bf16f32(a0, b, accD[s]);
                accD[s + 4] = wmma_bf16f32(a1, b, accD[s + 4]);
            }
        }
        __syncthreads();
    }
    // ---------------- epilogue: normalize + scatter ----------------
    const float4* slo = (const float4*)&rowsum[hb * 8];
    const float4* shi = (const float4*)&rowsum[16 + hb * 8];
    float4 s0 = slo[0], s1 = slo[1], t0 = shi[0], t1 = shi[1];
    float invLo[8] = {1.0f / s0.x, 1.0f / s0.y, 1.0f / s0.z, 1.0f / s0.w,
                      1.0f / s1.x, 1.0f / s1.y, 1.0f / s1.z, 1.0f / s1.w};
    float invHi[8] = {1.0f / t0.x, 1.0f / t0.y, 1.0f / t0.z, 1.0f / t0.w,
                      1.0f / t1.x, 1.0f / t1.y, 1.0f / t1.z, 1.0f / t1.w};
#pragma unroll
    for (int r = 0; r < 8; ++r) {
        int m = r + 8 * hb;
        int rI0 = sperm[m], rI1 = sperm[16 + m];
        if (rI0 >= 0) {
#pragma unroll
            for (int s = 0; s < 4; ++s)
                Out[(size_t)rI0 * DDIM + w * 64 + s * 16 + mn] = accD[s][r] * invLo[r];
        }
        if (rI1 >= 0) {
#pragma unroll
            for (int s = 0; s < 4; ++s)
                Out[(size_t)rI1 * DDIM + w * 64 + s * 16 + mn] = accD[s + 4][r] * invHi[r];
        }
    }
}

// ---------------------------------------------------------------------------
// Host launcher. ws layout (256B-aligned):
//   perm [0,9216)  abase [9216,9252)  xin [9472,+640KB)  h [+4MB)  t [+4MB)
// ---------------------------------------------------------------------------
extern "C" void kernel_launch(void* const* d_in, const int* in_sizes, int n_in,
                              void* d_out, int out_size, void* d_ws, size_t ws_size,
                              hipStream_t stream) {
    const float* z   = (const float*)d_in[0];
    const int*   cid = (const int*)  d_in[1];
    const float* W1  = (const float*)d_in[2];
    const float* b1  = (const float*)d_in[3];
    const float* W2  = (const float*)d_in[4];
    const float* b2  = (const float*)d_in[5];
    const float* Wa  = (const float*)d_in[6];
    const float* ba  = (const float*)d_in[7];
    const float* Xb  = (const float*)d_in[8];
    float* out = (float*)d_out;

    char* ws = (char*)d_ws;
    int*            permp = (int*)(ws + 0);
    int*            abse  = (int*)(ws + 9216);
    unsigned short* xin   = (unsigned short*)(ws + 9472);
    unsigned short* h     = (unsigned short*)(ws + 9472 + 655360);
    unsigned short* t     = (unsigned short*)(ws + 9472 + 655360 + 4194304);

    k_sort<<<1, 256, 0, stream>>>(cid, permp, abse);
    k_xin<<<(BB * KPAD1 + 255) / 256, 256, 0, stream>>>(z, cid, xin);
    k_mlp<<<dim3(BB / 16, HID / 128), 256, 0, stream>>>(xin, KPAD1, W1, HID, LATENT + NCLS, b1, h, HID);
    k_mlp<<<dim3(BB / 16, HID / 128), 256, 0, stream>>>(h, HID, W2, HID, HID, b2, t, HID);
    k_attend<<<PADB / 32, 256, 0, stream>>>(t, Wa, ba, Xb, permp, abse, out);
}
